// ScaledDotProductAttention_43980465111324
// MI455X (gfx1250) — compile-verified, matrix operand order checked by reading, not run on hardware
//
#include <hip/hip_runtime.h>

typedef __attribute__((ext_vector_type(16))) _Float16 v16h;
typedef __attribute__((ext_vector_type(8)))  _Float16 v8h;
typedef __attribute__((ext_vector_type(4)))  _Float16 v4h;
typedef __attribute__((ext_vector_type(8)))  float    v8f;
typedef __attribute__((ext_vector_type(4)))  int      v4i;

// Problem constants (reference: B,H,S,D = 4,16,2048,64)
constexpr int Bn = 4, Hn = 16, Sn = 2048, Dn = 64;
constexpr int QTILE = 128;     // queries per workgroup (8 waves x 16 rows)
constexpr int KBLK  = 128;     // keys per LDS block (8 score tiles)
constexpr int NBLK  = Sn / KBLK;
constexpr int NT    = KBLK / 16;   // score tiles per block = 8
// padded LDS row strides (halves): stride mod 64 banks -> conflict-free b128
constexpr int KD = Dn + 8;     // 72  (K tile row = 64 d's)
constexpr int VD = KBLK + 8;   // 136 (V^T tile row = 128 keys)
constexpr int PD = KBLK + 8;   // 136 (P staging row = 128 keys)

#if defined(__has_builtin)
#if __has_builtin(__builtin_amdgcn_global_load_async_to_lds_b128)
#define HAVE_ASYNC_LDS 1
#endif
#endif
#ifndef HAVE_ASYNC_LDS
#define HAVE_ASYNC_LDS 0
#endif

#define AS_GLOBAL __attribute__((address_space(1)))
#define AS_LDS    __attribute__((address_space(3)))

// 2^x via v_exp_f32 (hardware exp is base-2)
__device__ inline float fast_exp2(float x) {
#if defined(__has_builtin) && __has_builtin(__builtin_amdgcn_exp2f)
    return __builtin_amdgcn_exp2f(x);
#else
    return __expf(x * 0.6931471805599453f);
#endif
}

__device__ inline float fast_rcp(float x) {
#if defined(__has_builtin) && __has_builtin(__builtin_amdgcn_rcpf)
    return __builtin_amdgcn_rcpf(x);
#else
    return 1.0f / x;
#endif
}

__device__ inline void wait_async_lds() {
#if HAVE_ASYNC_LDS
#if defined(__has_builtin) && __has_builtin(__builtin_amdgcn_s_wait_asynccnt)
    __builtin_amdgcn_s_wait_asynccnt(0);
#else
    asm volatile("s_wait_asynccnt 0x0" ::: "memory");
#endif
#endif
}

// Copy one 16-byte chunk global(f16) -> LDS, async if available.
__device__ inline void copy16_to_lds(const _Float16* g, _Float16* l) {
#if HAVE_ASYNC_LDS
    __builtin_amdgcn_global_load_async_to_lds_b128(
        (AS_GLOBAL v4i*)g, (AS_LDS v4i*)l, 0, 0);
#else
    *(v8h*)l = *(const v8h*)g;
#endif
}

// Load a v16h WMMA operand from LDS following the 16-bit A/B register layout:
// element j in [0,7]  -> K = base + j        (contiguous)
// element j in [8,15] -> K = base + 16 + j-8 (contiguous)
__device__ inline v16h ld_op16(const _Float16* p) {
    union { v16h v; v8h h[2]; } u;
    u.h[0] = *(const v8h*)(p);
    u.h[1] = *(const v8h*)(p + 16);
    return u.v;
}

// Load a Q A-operand chunk from global fp32 with scale folded in.
__device__ inline v16h ld_q(const float* p, float scale) {
    v16h r;
#pragma unroll
    for (int j = 0; j < 8; ++j) {
        r[j]     = (_Float16)(p[j]      * scale);
        r[j + 8] = (_Float16)(p[16 + j] * scale);
    }
    return r;
}

// ---------------- pre-pass: K -> f16 [bh][s][d], V -> f16 transposed [bh][d][s]
__global__ __launch_bounds__(256)
void convert_kv_kernel(const float* __restrict__ K, const float* __restrict__ V,
                       _Float16* __restrict__ Kh, _Float16* __restrict__ Vt) {
    const size_t idx4 = (size_t)blockIdx.x * 256 + threadIdx.x; // float4 index
    const size_t e    = idx4 * 4;                               // element index

    float4 kf = ((const float4*)K)[idx4];
    v4h kh;
    kh[0] = (_Float16)kf.x; kh[1] = (_Float16)kf.y;
    kh[2] = (_Float16)kf.z; kh[3] = (_Float16)kf.w;
    *(v4h*)&Kh[e] = kh;

    float4 vf = ((const float4*)V)[idx4];
    const size_t bh  = e / ((size_t)Sn * Dn);
    const int    rem = (int)(e - bh * (size_t)Sn * Dn);
    const int    s   = rem / Dn;
    const int    d   = rem % Dn;
    _Float16* vt = Vt + bh * (size_t)Sn * Dn;
    vt[(size_t)(d + 0) * Sn + s] = (_Float16)vf.x;
    vt[(size_t)(d + 1) * Sn + s] = (_Float16)vf.y;
    vt[(size_t)(d + 2) * Sn + s] = (_Float16)vf.z;
    vt[(size_t)(d + 3) * Sn + s] = (_Float16)vf.w;
}

// ---------------- main flash-attention kernel ----------------
__global__ __launch_bounds__(256)
void fa_wmma_kernel(const float* __restrict__ Q,
                    const _Float16* __restrict__ Kh,   // [bh][s][d] f16
                    const _Float16* __restrict__ Vt,   // [bh][d][s] f16
                    float* __restrict__ O) {
    // Double-buffered K/V tiles (padded rows) + per-wave P staging.
    __shared__ __align__(16) _Float16 Ksh[2][KBLK * KD];   // 2 x 18 KB
    __shared__ __align__(16) _Float16 Vts[2][Dn * VD];     // 2 x 17 KB
    __shared__ __align__(16) _Float16 Pst[8 * 16 * PD];    // 34 KB

    const int qb   = blockIdx.x;
    const int h    = blockIdx.y;
    const int bb   = blockIdx.z;
    const int tid  = threadIdx.x;
    const int lane = tid & 31;
    const int wv   = tid >> 5;       // wave id 0..7
    const int hf   = lane >> 4;      // half-wave: 0 or 1
    const int l16  = lane & 15;
    const int koff = hf * 8;         // K-offset pattern per half-wave

    const size_t bh = ((size_t)bb * Hn + h) * (size_t)Sn * Dn;
    const float*    Qb  = Q  + bh;
    const _Float16* KhB = Kh + bh;
    const _Float16* VtB = Vt + bh;
    float*          Ob  = O  + bh;

    // fold softmax scale and log2(e) into Q so scores are in exp2 domain
    const float qscale = 0.125f * 1.44269504088896f;
    const int   qrow   = qb * QTILE + wv * 16 + l16;

    v16h qa0 = ld_q(Qb + (size_t)qrow * Dn + 0 * 32 + koff, qscale);
    v16h qa1 = ld_q(Qb + (size_t)qrow * Dn + 1 * 32 + koff, qscale);

    v16h vones;
#pragma unroll
    for (int j = 0; j < 16; ++j) vones[j] = (_Float16)1.0f;

    const v8f vzero = {0.f, 0.f, 0.f, 0.f, 0.f, 0.f, 0.f, 0.f};
    v8f o[4];
#pragma unroll
    for (int t = 0; t < 4; ++t) o[t] = vzero;
    v8f olsum = vzero;               // row sums of P, accumulated by WMMA

    float m[8];
#pragma unroll
    for (int v = 0; v < 8; ++v) m[v] = -1e30f;

    _Float16* Pw = &Pst[wv * 16 * PD];

    // Fill tile kb into buffer buf: 2048 16B-chunks total, 8 per thread.
    auto fill_tile = [&](int kb, int buf) {
#pragma unroll
        for (int i = 0; i < 4; ++i) {
            const int c = tid + i * 256;          // chunk id 0..1023
            // K tile: 128 keys x 64 d -> 8 chunks per row
            {
                const int row = c >> 3, cc = c & 7;
                copy16_to_lds(KhB + (size_t)kb * KBLK * Dn + row * Dn + cc * 8,
                              &Ksh[buf][row * KD + cc * 8]);
            }
            // V^T tile: 64 d-rows x 128 keys -> 16 chunks per row
            {
                const int row = c >> 4, cc = c & 15;
                copy16_to_lds(VtB + (size_t)row * Sn + kb * KBLK + cc * 8,
                              &Vts[buf][row * VD + cc * 8]);
            }
        }
    };

    fill_tile(0, 0);   // preload first tile

    for (int kb = 0; kb < NBLK; ++kb) {
        const int cur = kb & 1;
        wait_async_lds();        // this wave's async fills complete
        __syncthreads();         // everyone's fills complete / prev compute done
        if (kb + 1 < NBLK) fill_tile(kb + 1, cur ^ 1);  // overlap with compute

        // ---- S = Q * K^T : 8 key tiles of 16, K-dim D=64 in 2 chunks ----
        v8f c[NT];
#pragma unroll
        for (int t = 0; t < NT; ++t) c[t] = vzero;
#pragma unroll
        for (int kc = 0; kc < 2; ++kc) {
            v16h a = kc ? qa1 : qa0;
#pragma unroll
            for (int t = 0; t < NT; ++t) {
                v16h bop = ld_op16(&Ksh[cur][(t * 16 + l16) * KD + kc * 32 + koff]);
                c[t] = __builtin_amdgcn_wmma_f32_16x16x32_f16(
                    false, a, false, bop, (short)0, c[t], false, false);
            }
        }

        // ---- online softmax in exp2 domain (row max via half-wave shuffle) ----
        float sc[8];
#pragma unroll
        for (int v = 0; v < 8; ++v) {
            float r = fmaxf(fmaxf(fmaxf(c[0][v], c[1][v]), fmaxf(c[2][v], c[3][v])),
                            fmaxf(fmaxf(c[4][v], c[5][v]), fmaxf(c[6][v], c[7][v])));
            r = fmaxf(r, __shfl_xor(r, 1, 32));
            r = fmaxf(r, __shfl_xor(r, 2, 32));
            r = fmaxf(r, __shfl_xor(r, 4, 32));
            r = fmaxf(r, __shfl_xor(r, 8, 32));
            const float mn = fmaxf(m[v], r);
            sc[v] = fast_exp2(m[v] - mn);
            m[v]  = mn;
        }
#pragma unroll
        for (int t = 0; t < NT; ++t)
#pragma unroll
            for (int v = 0; v < 8; ++v)
                c[t][v] = fast_exp2(c[t][v] - m[v]);

        // rescale accumulators (incl. row-sum accumulator)
#pragma unroll
        for (int v = 0; v < 8; ++v) {
#pragma unroll
            for (int t = 0; t < 4; ++t) o[t][v] *= sc[v];
            olsum[v] *= sc[v];
        }

        // ---- stage P (C-layout -> row-major) for A-operand reload ----
#pragma unroll
        for (int t = 0; t < NT; ++t)
#pragma unroll
            for (int v = 0; v < 8; ++v)
                Pw[(v + 8 * hf) * PD + t * 16 + l16] = (_Float16)c[t][v];

        // ---- O += P*V, rowsum += P*ones : K-dim = 128 keys in 4 chunks ----
#pragma unroll
        for (int kc = 0; kc < 4; ++kc) {
            v16h pa = ld_op16(&Pw[l16 * PD + kc * 32 + koff]);
            olsum = __builtin_amdgcn_wmma_f32_16x16x32_f16(
                false, pa, false, vones, (short)0, olsum, false, false);
#pragma unroll
            for (int t = 0; t < 4; ++t) {
                v16h vb = ld_op16(&Vts[cur][(t * 16 + l16) * VD + kc * 32 + koff]);
                o[t] = __builtin_amdgcn_wmma_f32_16x16x32_f16(
                    false, pa, false, vb, (short)0, o[t], false, false);
            }
        }
    }

    // ---- normalize (1 rcp per row) and write out (fp32) ----
#pragma unroll
    for (int v = 0; v < 8; ++v) {
        const float rinv = fast_rcp(olsum[v]);
        const int   row  = qb * QTILE + wv * 16 + v + 8 * hf;
#pragma unroll
        for (int t = 0; t < 4; ++t)
            Ob[(size_t)row * Dn + t * 16 + l16] = o[t][v] * rinv;
    }
}

extern "C" void kernel_launch(void* const* d_in, const int* in_sizes, int n_in,
                              void* d_out, int out_size, void* d_ws, size_t ws_size,
                              hipStream_t stream) {
    const float* q = (const float*)d_in[0];
    const float* k = (const float*)d_in[1];
    const float* v = (const float*)d_in[2];
    float* out = (float*)d_out;
    (void)in_sizes; (void)n_in; (void)out_size; (void)ws_size;

    const size_t elems = (size_t)Bn * Hn * Sn * Dn;      // 8,388,608
    _Float16* Kh = (_Float16*)d_ws;                      // 16 MB
    _Float16* Vt = Kh + elems;                           // 16 MB

    // Pre-pass: fp32 -> f16, V transposed per (b,h). 4 elems/thread.
    const int cgrid = (int)(elems / (256 * 4));          // 8192 blocks
    convert_kv_kernel<<<cgrid, 256, 0, stream>>>(k, v, Kh, Vt);

    dim3 grid(Sn / QTILE, Hn, Bn);                       // (16, 16, 4)
    fa_wmma_kernel<<<grid, 256, 0, stream>>>(q, Kh, Vt, out);
}
